// Task2Net_86277303042078
// MI455X (gfx1250) — matrix operand order repaired and checked by reading, not compiled, not run on hardware
//
#include <hip/hip_runtime.h>
#include <cstdint>
#include <cstddef>

// ---------------------------------------------------------------------------
// CDNA5 WMMA types / helpers (gfx1250, wave32)
// ---------------------------------------------------------------------------
typedef __attribute__((ext_vector_type(16))) __bf16 v16bf;
typedef __attribute__((ext_vector_type(8)))  float  v8f;

union Frag16 { v16bf v; uint4 q[2]; unsigned short s[16]; };
union Acc8   { v8f v; float f[8]; };

__device__ __forceinline__ unsigned short f2bf(float x) {
  union { float f; unsigned u; } t; t.f = x;
  unsigned r = t.u + 0x7fffu + ((t.u >> 16) & 1u);   // round-to-nearest-even
  return (unsigned short)(r >> 16);
}

#define WMMA_BF16(A_, B_, C_) \
  __builtin_amdgcn_wmma_f32_16x16x32_bf16(false, (A_), false, (B_), (short)0, (C_), false, false)

// ---------------------------------------------------------------------------
// CDNA5 async global->LDS copy (ASYNCcnt-tracked; see CDNA5 ISA)
// Generic LDS pointers on AMDGPU are {shared_aperture_hi32, lds_offset}:
// low 32 bits give the DS byte address the instruction needs.
// ---------------------------------------------------------------------------
__device__ __forceinline__ void async_load_b128(unsigned lds_byte_addr, const void* gaddr) {
  asm volatile("global_load_async_to_lds_b128 %0, %1, off"
               :: "v"(lds_byte_addr), "v"(gaddr) : "memory");
}
__device__ __forceinline__ void wait_async0() {
  asm volatile("s_wait_asynccnt 0x0" ::: "memory");
}
__device__ __forceinline__ unsigned lds_addr_of(const void* p) {
  return (unsigned)(uintptr_t)p;
}

// ---------------------------------------------------------------------------
// Small utility kernels
// ---------------------------------------------------------------------------
__global__ void k_zero_f32(float* __restrict__ p, long n) {
  long i = (long)blockIdx.x * blockDim.x + threadIdx.x;
  if (i < n) p[i] = 0.0f;
}

__global__ void k_degrees(const int* __restrict__ src, const int* __restrict__ dst,
                          float* __restrict__ degO, float* __restrict__ degI, int E) {
  int e = blockIdx.x * blockDim.x + threadIdx.x;
  if (e < E) {
    unsafeAtomicAdd(&degO[src[e]], 1.0f);
    unsafeAtomicAdd(&degI[dst[e]], 1.0f);
  }
}

__global__ void k_invsqrt(float* __restrict__ a, float* __restrict__ b, int n) {
  int i = blockIdx.x * blockDim.x + threadIdx.x;
  if (i < n) {
    a[i] = rsqrtf(fmaxf(a[i], 1.0f));
    b[i] = rsqrtf(fmaxf(b[i], 1.0f));
  }
}

// W stored [K][N] row-major (x @ W convention) -> bf16 [Np][Kp] (row n contiguous in K)
__global__ void k_w_tr(const float* __restrict__ W, unsigned short* __restrict__ o,
                       int K, int Nn, int Kp, int Np) {
  int i = blockIdx.x * blockDim.x + threadIdx.x;
  if (i < Np * Kp) {
    int n = i / Kp, k = i - n * Kp;
    o[i] = (n < Nn && k < K) ? f2bf(W[(size_t)k * Nn + n]) : (unsigned short)0;
  }
}

// W stored [N][K] row-major (x @ W.T convention) -> bf16 [Np][Kp]
__global__ void k_w_nk(const float* __restrict__ W, unsigned short* __restrict__ o,
                       int Nn, int K, int Kp, int Np) {
  int i = blockIdx.x * blockDim.x + threadIdx.x;
  if (i < Np * Kp) {
    int n = i / Kp, k = i - n * Kp;
    o[i] = (n < Nn && k < K) ? f2bf(W[(size_t)n * K + k]) : (unsigned short)0;
  }
}

__global__ void k_scatter(const float* __restrict__ h, const int* __restrict__ src,
                          const int* __restrict__ dst, float* __restrict__ agg,
                          int E, int D) {
  long i = (long)blockIdx.x * blockDim.x + threadIdx.x;
  if (i < (long)E * D) {
    int e = (int)(i / D), d = (int)(i - (long)e * D);
    unsafeAtomicAdd(&agg[(size_t)dst[e] * 64 + d], h[(size_t)src[e] * 64 + d]);
  }
}

__global__ void k_gc_fin(const float* __restrict__ agg, const float* __restrict__ nd,
                         const float* __restrict__ bias, float* __restrict__ item,
                         int colOff, int D, int N) {
  long i = (long)blockIdx.x * blockDim.x + threadIdx.x;
  if (i < (long)N * D) {
    int r = (int)(i / D), d = (int)(i - (long)r * D);
    float v = agg[(size_t)r * 64 + d] * nd[r] + bias[d];
    v = (v > 0.0f) ? v : (expf(v) - 1.0f);            // ELU
    item[(size_t)r * 144 + colOff + d] = v;
  }
}

__global__ void k_item_bf16(const float* __restrict__ item, unsigned short* __restrict__ o,
                            int N, int NP) {
  long i = (long)blockIdx.x * blockDim.x + threadIdx.x;
  if (i < (long)NP * 160) {
    int r = (int)(i / 160), c = (int)(i - (long)r * 160);
    o[i] = (r < N && c < 144) ? f2bf(item[(size_t)r * 144 + c]) : (unsigned short)0;
  }
}

__global__ void k_gather_q(const float* __restrict__ item, const int* __restrict__ qi,
                           unsigned short* __restrict__ qb, int S) {
  long i = (long)blockIdx.x * blockDim.x + threadIdx.x;
  if (i < (long)S * 160) {
    int s = (int)(i / 160), c = (int)(i - (long)s * 160);
    int node = qi[s];
    qb[i] = (c < 144) ? f2bf(item[(size_t)node * 144 + c]) : (unsigned short)0;
  }
}

__global__ void k_gather_pn(const float* __restrict__ item, const int* __restrict__ pos,
                            const int* __restrict__ neg, float* __restrict__ op,
                            float* __restrict__ on, int B) {
  int i = blockIdx.x * blockDim.x + threadIdx.x;
  if (i < B * 144) {
    int b = i / 144, c = i - b * 144;
    op[i] = item[(size_t)pos[b] * 144 + c];
    on[i] = item[(size_t)neg[b] * 144 + c];
  }
}

__global__ void k_qemb_bf16(const float* __restrict__ qe, unsigned short* __restrict__ o, int B) {
  int i = blockIdx.x * blockDim.x + threadIdx.x;
  if (i < B * 160) {
    int r = i / 160, c = i - r * 160;
    o[i] = (c < 144) ? f2bf(qe[(size_t)r * 144 + c]) : (unsigned short)0;
  }
}

// ---------------------------------------------------------------------------
// Generic WMMA GEMM:  Y[M,Nn] = (A * rowScale) @ Wt^T   (Wt is bf16 [Np][Kp])
//   ABF16=false: A is f32 [M][lda], converted+scaled on the fly
//   ABF16=true : A is bf16 [Mpad][lda(=Kp)] already padded
//   EPI: 0 plain store, 1 bias + ELU
//   WPB waves per block; wave w owns N-tile blockIdx.y*WPB + w (Wt rows must
//   be padded to cover the widened grid).
// ---------------------------------------------------------------------------
template<bool ABF16, int EPI, int WPB>
__global__ __launch_bounds__(32 * WPB) void k_wmma_gemm(
    const void* __restrict__ Aptr, int lda,
    const float* __restrict__ rowScale,
    const unsigned short* __restrict__ Wt, int Kp,
    const float* __restrict__ bias,
    float* __restrict__ Y, long ldy,
    int M, int Nn, int K)
{
  const int lane  = threadIdx.x & 31;
  const int wv    = threadIdx.x >> 5;
  const int m0    = blockIdx.x * 16;
  const int n0    = (blockIdx.y * WPB + wv) * 16;
  const int arow  = m0 + (lane & 15);
  const int koff  = (lane >> 4) * 8;       // A: lanes 0-15 K[0..7,16..23]; 16-31 K[8..15,24..31]
  const int brow  = n0 + (lane & 15);      // B col n = weight row n
  const int khalf = (lane >> 4) * 8;       // B: lanes 0-15 K[0..7,16..23]; 16-31 K[8..15,24..31]

  Acc8 acc;
#pragma unroll
  for (int r = 0; r < 8; ++r) acc.f[r] = 0.0f;

  for (int kt = 0; kt < Kp; kt += 32) {
    Frag16 a, b;
    if constexpr (ABF16) {
      const unsigned short* A  = (const unsigned short*)Aptr;
      const unsigned short* ap = A + (size_t)arow * lda + kt;
      a.q[0] = *(const uint4*)(ap + koff);
      a.q[1] = *(const uint4*)(ap + koff + 16);
    } else {
      const float* A  = (const float*)Aptr;
      const bool   rok = arow < M;
      const float  scl = (rowScale && rok) ? rowScale[arow] : 1.0f;
      const float* ap  = A + (size_t)arow * lda;
#pragma unroll
      for (int i = 0; i < 8; ++i) {
        int k0 = kt + koff + i;
        int k1 = k0 + 16;
        float v0 = (rok && k0 < K) ? ap[k0] : 0.0f;
        float v1 = (rok && k1 < K) ? ap[k1] : 0.0f;
        a.s[i]     = f2bf(v0 * scl);
        a.s[8 + i] = f2bf(v1 * scl);
      }
    }
    const unsigned short* bp = Wt + (size_t)brow * Kp + kt + khalf;
    __builtin_prefetch(bp + 32, 0, 1);
    b.q[0] = *(const uint4*)(bp);
    b.q[1] = *(const uint4*)(bp + 16);
    acc.v = WMMA_BF16(a.v, b.v, acc.v);
  }

  const int mb  = (lane >> 4) * 8;
  const int col = n0 + (lane & 15);
#pragma unroll
  for (int r = 0; r < 8; ++r) {
    int mm = m0 + mb + r;
    if (mm < M && col < Nn) {
      float v = acc.f[r];
      if (EPI == 1) {
        v += bias[col];
        v = (v > 0.0f) ? v : (expf(v) - 1.0f);
      }
      Y[(size_t)mm * ldy + col] = v;
    }
  }
}

// ---------------------------------------------------------------------------
// Forward LSTM with projection. One block = 16 batch rows, 8 waves, all T
// steps locally. x tiles are staged into LDS with async global->LDS copies.
// gates[16,576] = x[16,160] @ Wih^T + h[16,96] @ Whh^T + (bih+bhh)
// ---------------------------------------------------------------------------
__global__ __launch_bounds__(256) void k_lstm_fwd(
    const unsigned short* __restrict__ qbf,   // [B*T][160] bf16
    const int* __restrict__ lengths,
    const unsigned short* __restrict__ Wih,   // [576][160] bf16
    const unsigned short* __restrict__ Whh,   // [576][96]  bf16
    const unsigned short* __restrict__ Whr,   // [80][160]  bf16
    const float* __restrict__ bih, const float* __restrict__ bhh,
    float* __restrict__ out)                  // query_embeds base, ld 144
{
  __shared__ __attribute__((aligned(16))) unsigned short xs[16 * 160];
  __shared__ __attribute__((aligned(16))) unsigned short us[16 * 160];
  __shared__ __attribute__((aligned(16))) unsigned short hs[16 * 96];
  __shared__ float gs[16 * 576];
  __shared__ float cs[16 * 144];
  __shared__ float hf[16 * 72];
  __shared__ int   lenS[16];

  const int tid   = threadIdx.x;
  const int wv    = tid >> 5;
  const int lane  = tid & 31;
  const int b0    = blockIdx.x * 16;
  const int arow  = lane & 15;
  const int koff  = (lane >> 4) * 8;
  const int khalf = (lane >> 4) * 8;
  const unsigned xsA = lds_addr_of(xs);

  for (int i = tid; i < 16 * 160; i += 256) us[i] = 0;
  for (int i = tid; i < 16 * 96;  i += 256) hs[i] = 0;
  for (int i = tid; i < 16 * 144; i += 256) cs[i] = 0.0f;
  for (int i = tid; i < 16 * 72;  i += 256) hf[i] = 0.0f;
  if (tid < 16) lenS[tid] = lengths[b0 + tid];
  __syncthreads();

  for (int t = 0; t < 50; ++t) {
    // stage x_t tile (16 rows x 160 bf16 = 320 x b128) via async global->LDS
    for (int i = tid; i < 16 * 20; i += 256) {
      int r = i / 20, c = i - r * 20;
      async_load_b128(xsA + (unsigned)(r * 320 + c * 16),
                      qbf + ((size_t)(b0 + r) * 50 + t) * 160 + c * 8);
    }
    wait_async0();
    __syncthreads();

    // gates GEMM: 36 N-tiles over 8 waves
    for (int nt = wv; nt < 36; nt += 8) {
      const int n0t  = nt * 16;
      const int colg = n0t + (lane & 15);
      const float bval = bih[colg] + bhh[colg];
      Acc8 acc;
#pragma unroll
      for (int r = 0; r < 8; ++r) acc.f[r] = bval;
#pragma unroll
      for (int kt = 0; kt < 160; kt += 32) {               // x @ Wih^T
        Frag16 a, b;
        a.q[0] = *(const uint4*)(xs + arow * 160 + kt + koff);
        a.q[1] = *(const uint4*)(xs + arow * 160 + kt + koff + 16);
        const unsigned short* bp = Wih + (size_t)colg * 160 + kt + khalf;
        b.q[0] = *(const uint4*)(bp);
        b.q[1] = *(const uint4*)(bp + 16);
        acc.v = WMMA_BF16(a.v, b.v, acc.v);
      }
#pragma unroll
      for (int kt = 0; kt < 96; kt += 32) {                // h @ Whh^T
        Frag16 a, b;
        a.q[0] = *(const uint4*)(hs + arow * 96 + kt + koff);
        a.q[1] = *(const uint4*)(hs + arow * 96 + kt + koff + 16);
        const unsigned short* bp = Whh + (size_t)colg * 96 + kt + khalf;
        b.q[0] = *(const uint4*)(bp);
        b.q[1] = *(const uint4*)(bp + 16);
        acc.v = WMMA_BF16(a.v, b.v, acc.v);
      }
#pragma unroll
      for (int r = 0; r < 8; ++r) {
        int mm = (lane >> 4) * 8 + r;
        gs[mm * 576 + colg] = acc.f[r];
      }
    }
    __syncthreads();

    // nonlinearity + cell update (i,f,g,o gate order)
    for (int i = tid; i < 16 * 144; i += 256) {
      int m = i / 144, j = i - m * 144;
      float gi = gs[m * 576 + j];
      float gf = gs[m * 576 + 144 + j];
      float gg = gs[m * 576 + 288 + j];
      float go = gs[m * 576 + 432 + j];
      float si = 1.0f / (1.0f + expf(-gi));
      float sf = 1.0f / (1.0f + expf(-gf));
      float tg = tanhf(gg);
      float so = 1.0f / (1.0f + expf(-go));
      float cn = sf * cs[i] + si * tg;
      us[m * 160 + j] = f2bf(so * tanhf(cn));
      if (t < lenS[m]) cs[i] = cn;                 // masked c commit
    }
    __syncthreads();

    // projection: h_new[16,72] = u[16,160] @ Whr^T ; 5 N-tiles on waves 0-4
    if (wv < 5) {
      const int n0t  = wv * 16;
      const int browp = n0t + (lane & 15);
      Acc8 acc;
#pragma unroll
      for (int r = 0; r < 8; ++r) acc.f[r] = 0.0f;
#pragma unroll
      for (int kt = 0; kt < 160; kt += 32) {
        Frag16 a, b;
        a.q[0] = *(const uint4*)(us + arow * 160 + kt + koff);
        a.q[1] = *(const uint4*)(us + arow * 160 + kt + koff + 16);
        const unsigned short* bp = Whr + (size_t)browp * 160 + kt + khalf;
        b.q[0] = *(const uint4*)(bp);
        b.q[1] = *(const uint4*)(bp + 16);
        acc.v = WMMA_BF16(a.v, b.v, acc.v);
      }
#pragma unroll
      for (int r = 0; r < 8; ++r) {
        int mm  = (lane >> 4) * 8 + r;
        int col = n0t + (lane & 15);
        if (col < 72 && t < lenS[mm]) {            // masked h commit
          hf[mm * 72 + col] = acc.f[r];
          hs[mm * 96 + col] = f2bf(acc.f[r]);
        }
      }
    }
    __syncthreads();
  }

  for (int i = tid; i < 16 * 72; i += 256) {
    int m = i / 72, n = i - m * 72;
    out[(size_t)(b0 + m) * 144 + n] = hf[i];       // h_fwd -> cols 0..71
  }
}

// ---------------------------------------------------------------------------
// Backward direction: single cell step on x at t = len-1, zero initial state.
// ---------------------------------------------------------------------------
__global__ __launch_bounds__(256) void k_lstm_bwd(
    const unsigned short* __restrict__ qbf,
    const int* __restrict__ lengths,
    const unsigned short* __restrict__ Wih,
    const unsigned short* __restrict__ Whr,
    const float* __restrict__ bih, const float* __restrict__ bhh,
    float* __restrict__ out)
{
  __shared__ __attribute__((aligned(16))) unsigned short xs[16 * 160];
  __shared__ __attribute__((aligned(16))) unsigned short us[16 * 160];
  __shared__ float gs[16 * 576];

  const int tid   = threadIdx.x;
  const int wv    = tid >> 5;
  const int lane  = tid & 31;
  const int b0    = blockIdx.x * 16;
  const int arow  = lane & 15;
  const int koff  = (lane >> 4) * 8;
  const int khalf = (lane >> 4) * 8;
  const unsigned xsA = lds_addr_of(xs);

  for (int i = tid; i < 16 * 160; i += 256) us[i] = 0;
  for (int i = tid; i < 16 * 20; i += 256) {
    int r = i / 20, c = i - r * 20;
    int L = lengths[b0 + r];
    async_load_b128(xsA + (unsigned)(r * 320 + c * 16),
                    qbf + ((size_t)(b0 + r) * 50 + (L - 1)) * 160 + c * 8);
  }
  wait_async0();
  __syncthreads();

  for (int nt = wv; nt < 36; nt += 8) {
    const int n0t  = nt * 16;
    const int colg = n0t + (lane & 15);
    const float bval = bih[colg] + bhh[colg];
    Acc8 acc;
#pragma unroll
    for (int r = 0; r < 8; ++r) acc.f[r] = bval;
#pragma unroll
    for (int kt = 0; kt < 160; kt += 32) {
      Frag16 a, b;
      a.q[0] = *(const uint4*)(xs + arow * 160 + kt + koff);
      a.q[1] = *(const uint4*)(xs + arow * 160 + kt + koff + 16);
      const unsigned short* bp = Wih + (size_t)colg * 160 + kt + khalf;
      b.q[0] = *(const uint4*)(bp);
      b.q[1] = *(const uint4*)(bp + 16);
      acc.v = WMMA_BF16(a.v, b.v, acc.v);
    }
#pragma unroll
    for (int r = 0; r < 8; ++r) {
      int mm = (lane >> 4) * 8 + r;
      gs[mm * 576 + colg] = acc.f[r];
    }
  }
  __syncthreads();

  for (int i = tid; i < 16 * 144; i += 256) {
    int m = i / 144, j = i - m * 144;
    float gi = gs[m * 576 + j];
    float gg = gs[m * 576 + 288 + j];
    float go = gs[m * 576 + 432 + j];
    float si = 1.0f / (1.0f + expf(-gi));
    float tg = tanhf(gg);
    float so = 1.0f / (1.0f + expf(-go));
    float c  = si * tg;                         // c0 = 0, no forget term
    us[m * 160 + j] = f2bf(so * tanhf(c));
  }
  __syncthreads();

  if (wv < 5) {
    const int n0t  = wv * 16;
    const int browp = n0t + (lane & 15);
    Acc8 acc;
#pragma unroll
    for (int r = 0; r < 8; ++r) acc.f[r] = 0.0f;
#pragma unroll
    for (int kt = 0; kt < 160; kt += 32) {
      Frag16 a, b;
      a.q[0] = *(const uint4*)(us + arow * 160 + kt + koff);
      a.q[1] = *(const uint4*)(us + arow * 160 + kt + koff + 16);
      const unsigned short* bp = Whr + (size_t)browp * 160 + kt + khalf;
      b.q[0] = *(const uint4*)(bp);
      b.q[1] = *(const uint4*)(bp + 16);
      acc.v = WMMA_BF16(a.v, b.v, acc.v);
    }
#pragma unroll
    for (int r = 0; r < 8; ++r) {
      int mm  = (lane >> 4) * 8 + r;
      int col = n0t + (lane & 15);
      if (col < 72)
        out[(size_t)(b0 + mm) * 144 + 72 + col] = acc.f[r];   // h_bwd -> cols 72..143
    }
  }
}

// ---------------------------------------------------------------------------
// Host orchestration
// ---------------------------------------------------------------------------
extern "C" void kernel_launch(void* const* d_in, const int* in_sizes, int n_in,
                              void* d_out, int out_size, void* d_ws, size_t ws_size,
                              hipStream_t stream) {
  (void)in_sizes; (void)n_in; (void)out_size; (void)ws_size;
  const int N = 50000, E = 1600000, Bq = 1024, T = 50;
  const int NP  = 50016;   // f32 row pad (multiple of 16)
  const int NPB = 50048;   // bf16 row pad (multiple of 64 for 4-wave logit blocks)

  const float* node_feature = (const float*)d_in[0];
  const int*   src  = (const int*)d_in[1];
  const int*   dst  = (const int*)d_in[2];
  const int*   qidx = (const int*)d_in[3];
  const int*   pos  = (const int*)d_in[4];
  const int*   neg  = (const int*)d_in[5];
  const int*   lens = (const int*)d_in[6];
  const float* Wfc  = (const float*)d_in[7];
  const float* bfc  = (const float*)d_in[8];
  const float* W1   = (const float*)d_in[9];
  const float* b1   = (const float*)d_in[10];
  const float* W2   = (const float*)d_in[11];
  const float* b2   = (const float*)d_in[12];
  const float* W3   = (const float*)d_in[13];
  const float* b3   = (const float*)d_in[14];
  const float* WihF = (const float*)d_in[15];
  const float* WhhF = (const float*)d_in[16];
  const float* bihF = (const float*)d_in[17];
  const float* bhhF = (const float*)d_in[18];
  const float* WhrF = (const float*)d_in[19];
  const float* WihB = (const float*)d_in[20];
  /* d_in[21] = Whh_b unused: backward step starts from h0 == 0 */
  const float* bihB = (const float*)d_in[22];
  const float* bhhB = (const float*)d_in[23];
  const float* WhrB = (const float*)d_in[24];

  float* outQ = (float*)d_out;                 // [1024,144] query_embeds
  float* outP = outQ + (size_t)Bq * 144;       // [1024,144] pos_embed
  float* outN = outP + (size_t)Bq * 144;       // [1024,144] neg_embed
  float* outL = outN + (size_t)Bq * 144;       // [1024,50000] logit

  char* ws = (char*)d_ws;
  size_t off = 0;
  auto alloc = [&](size_t bytes) -> void* {
    void* p = ws + off; off += (bytes + 255) & ~(size_t)255; return p;
  };
  float* ns    = (float*)alloc((size_t)NP * 4);
  float* nd    = (float*)alloc((size_t)NP * 4);
  float* fbuf  = (float*)alloc((size_t)NP * 32 * 4);
  float* hpre  = (float*)alloc((size_t)NP * 64 * 4);
  float* agg   = (float*)alloc((size_t)NP * 64 * 4);
  float* item  = (float*)alloc((size_t)NP * 144 * 4);
  unsigned short* itemB = (unsigned short*)alloc((size_t)NPB * 160 * 2);
  unsigned short* qB    = (unsigned short*)alloc((size_t)Bq * T * 160 * 2);
  unsigned short* qeB   = (unsigned short*)alloc((size_t)Bq * 160 * 2);
  unsigned short* WfcT  = (unsigned short*)alloc((size_t)32 * 128 * 2);
  unsigned short* W1T   = (unsigned short*)alloc((size_t)32 * 32 * 2);
  unsigned short* W2T   = (unsigned short*)alloc((size_t)48 * 32 * 2);
  unsigned short* W3T   = (unsigned short*)alloc((size_t)64 * 64 * 2);
  unsigned short* WihFb = (unsigned short*)alloc((size_t)576 * 160 * 2);
  unsigned short* WhhFb = (unsigned short*)alloc((size_t)576 * 96 * 2);
  unsigned short* WhrFb = (unsigned short*)alloc((size_t)80 * 160 * 2);
  unsigned short* WihBb = (unsigned short*)alloc((size_t)576 * 160 * 2);
  unsigned short* WhrBb = (unsigned short*)alloc((size_t)80 * 160 * 2);

  auto cdiv = [](long a, long b) { return (unsigned)((a + b - 1) / b); };

  // degrees + symmetric norms
  k_zero_f32<<<cdiv(NP, 256), 256, 0, stream>>>(ns, NP);
  k_zero_f32<<<cdiv(NP, 256), 256, 0, stream>>>(nd, NP);
  k_degrees<<<cdiv(E, 256), 256, 0, stream>>>(src, dst, ns, nd, E);
  k_invsqrt<<<cdiv(NP, 256), 256, 0, stream>>>(ns, nd, NP);

  // bf16 weight prep ([K][N] -> transposed; [N][K] -> direct; all K padded)
  k_w_tr<<<cdiv(32 * 128, 256), 256, 0, stream>>>(Wfc, WfcT, 128, 32, 128, 32);
  k_w_tr<<<cdiv(32 * 32, 256), 256, 0, stream>>>(W1, W1T, 32, 32, 32, 32);
  k_w_tr<<<cdiv(48 * 32, 256), 256, 0, stream>>>(W2, W2T, 32, 48, 32, 48);
  k_w_tr<<<cdiv(64 * 64, 256), 256, 0, stream>>>(W3, W3T, 48, 64, 64, 64);
  k_w_nk<<<cdiv(576 * 160, 256), 256, 0, stream>>>(WihF, WihFb, 576, 144, 160, 576);
  k_w_nk<<<cdiv(576 * 96, 256), 256, 0, stream>>>(WhhF, WhhFb, 576, 72, 96, 576);
  k_w_nk<<<cdiv(80 * 160, 256), 256, 0, stream>>>(WhrF, WhrFb, 72, 144, 160, 80);
  k_w_nk<<<cdiv(576 * 160, 256), 256, 0, stream>>>(WihB, WihBb, 576, 144, 160, 576);
  k_w_nk<<<cdiv(80 * 160, 256), 256, 0, stream>>>(WhrB, WhrBb, 72, 144, 160, 80);

  // feature_fc : f = elu(node_feature @ Wfc + bfc)
  k_wmma_gemm<false, 1, 1><<<dim3(3125, 2), 32, 0, stream>>>(
      node_feature, 128, nullptr, WfcT, 128, bfc, fbuf, 32, N, 32, 128);

  // GraphConv 1 (32 -> 32), e1 -> item cols [0,32)
  k_zero_f32<<<cdiv((long)NP * 64, 256), 256, 0, stream>>>(agg, (long)NP * 64);
  k_wmma_gemm<false, 0, 1><<<dim3(3125, 2), 32, 0, stream>>>(
      fbuf, 32, ns, W1T, 32, nullptr, hpre, 64, N, 32, 32);
  k_scatter<<<cdiv((long)E * 32, 256), 256, 0, stream>>>(hpre, src, dst, agg, E, 32);
  k_gc_fin<<<cdiv((long)N * 32, 256), 256, 0, stream>>>(agg, nd, b1, item, 0, 32, N);

  // GraphConv 2 (32 -> 48), e2 -> item cols [32,80)
  k_zero_f32<<<cdiv((long)NP * 64, 256), 256, 0, stream>>>(agg, (long)NP * 64);
  k_wmma_gemm<false, 0, 1><<<dim3(3125, 3), 32, 0, stream>>>(
      item, 144, ns, W2T, 32, nullptr, hpre, 64, N, 48, 32);
  k_scatter<<<cdiv((long)E * 48, 256), 256, 0, stream>>>(hpre, src, dst, agg, E, 48);
  k_gc_fin<<<cdiv((long)N * 48, 256), 256, 0, stream>>>(agg, nd, b2, item, 32, 48, N);

  // GraphConv 3 (48 -> 64), e3 -> item cols [80,144)
  k_zero_f32<<<cdiv((long)NP * 64, 256), 256, 0, stream>>>(agg, (long)NP * 64);
  k_wmma_gemm<false, 0, 1><<<dim3(3125, 4), 32, 0, stream>>>(
      item + 32, 144, ns, W3T, 64, nullptr, hpre, 64, N, 64, 48);
  k_scatter<<<cdiv((long)E * 64, 256), 256, 0, stream>>>(hpre, src, dst, agg, E, 64);
  k_gc_fin<<<cdiv((long)N * 64, 256), 256, 0, stream>>>(agg, nd, b3, item, 80, 64, N);

  // bf16 copies + gathers
  k_item_bf16<<<cdiv((long)NPB * 160, 256), 256, 0, stream>>>(item, itemB, N, NPB);
  k_gather_q<<<cdiv((long)Bq * T * 160, 256), 256, 0, stream>>>(item, qidx, qB, Bq * T);
  k_gather_pn<<<cdiv((long)Bq * 144, 256), 256, 0, stream>>>(item, pos, neg, outP, outN, Bq);

  // bidirectional LSTM heads
  k_lstm_fwd<<<64, 256, 0, stream>>>(qB, lens, WihFb, WhhFb, WhrFb, bihF, bhhF, outQ);
  k_lstm_bwd<<<64, 256, 0, stream>>>(qB, lens, WihBb, WhrBb, bihB, bhhB, outQ);

  // logit = query_embeds @ item_embeds^T   (the 14.8 GFLOP GEMM)
  // 4 waves per block; 782*4 = 3128 N-tiles cover 50000 cols (itemB padded to 50048 rows)
  k_qemb_bf16<<<cdiv((long)Bq * 160, 256), 256, 0, stream>>>(outQ, qeB, Bq);
  k_wmma_gemm<true, 0, 4><<<dim3(64, 782), 128, 0, stream>>>(
      qeB, 160, nullptr, itemB, 160, nullptr, outL, 50000, Bq, N, 144);
}
